// NT_XentLoss_14370960572635
// MI455X (gfx1250) — compile-verified, matrix operand order checked by reading, not run on hardware
//
#include <hip/hip_runtime.h>
#include <hip/hip_bf16.h>

typedef __attribute__((ext_vector_type(16))) _Float16 v16h;
typedef __attribute__((ext_vector_type(8)))  _Float16 v8h;
typedef __attribute__((ext_vector_type(4)))  _Float16 v4h;
typedef __attribute__((ext_vector_type(8)))  float    v8f;

#define NROWS 8192           // 2N
#define DDIM  128            // feature dim
#define NSPLIT 8             // column splits (blockIdx.y)
#define COLS_PER_SPLIT (NROWS / NSPLIT)   // 1024
#define PANEL 128            // columns staged in LDS per iteration

__device__ __forceinline__ v16h cat8(v8h lo, v8h hi) {
    v16h r;
#pragma unroll
    for (int i = 0; i < 8; ++i) { r[i] = lo[i]; r[8 + i] = hi[i]; }
    return r;
}

// ---------------------------------------------------------------------------
// Kernel 1: interleave + L2-normalize rows, write f16 Zn[8192][128] to ws.
// One wave (32 lanes) per row; 8 waves per 256-thread block.
// ---------------------------------------------------------------------------
__launch_bounds__(256)
__global__ void normalize_rows(const float* __restrict__ zi,
                               const float* __restrict__ zj,
                               _Float16* __restrict__ zn) {
    const int wave = threadIdx.x >> 5;
    const int lane = threadIdx.x & 31;
    const int row  = blockIdx.x * 8 + wave;          // 0..8191
    const float* src = (row & 1) ? zj : zi;
    const float* p   = src + (size_t)(row >> 1) * DDIM;

    float4 x = ((const float4*)p)[lane];             // 32 lanes * 4 = 128
    float ss = x.x * x.x + x.y * x.y + x.z * x.z + x.w * x.w;
#pragma unroll
    for (int m = 16; m >= 1; m >>= 1) ss += __shfl_xor(ss, m, 32);
    const float rn = rsqrtf(ss);

    v4h o;
    o[0] = (_Float16)(x.x * rn);
    o[1] = (_Float16)(x.y * rn);
    o[2] = (_Float16)(x.z * rn);
    o[3] = (_Float16)(x.w * rn);
    *(v4h*)(zn + (size_t)row * DDIM + lane * 4) = o;
}

// ---------------------------------------------------------------------------
// Kernel 2: fused Gram + exp-row-sum + partner capture.
// grid = (64 row-blocks, 8 col-splits), block = 256 threads (8 waves).
// Wave w owns rows [rb*128 + w*16, +16). Block stages a 128-column panel of
// Zn in LDS, shared by all 8 waves; each 16x16 tile = 4x v_wmma_f32_16x16x32_f16.
// Diagonal/partner handling only occurs in the single tile with
// colBase == rowBase (16-aligned tiles; grow^1 stays in the same 16-row tile).
// That condition is wave-uniform, so we force it scalar via readfirstlane.
// ---------------------------------------------------------------------------
__launch_bounds__(256)
__global__ void gram_loss(const _Float16* __restrict__ zn,
                          float* __restrict__ den,
                          float* __restrict__ num) {
    __shared__ __align__(16) _Float16 Bs[PANEL * DDIM];   // 32 KB

    const int wave    = threadIdx.x >> 5;
    const int lane    = threadIdx.x & 31;
    const int halfsel = lane >> 4;          // 0: lanes 0-15, 1: lanes 16-31
    const int lm      = lane & 15;
    const int rowBase = blockIdx.x * 128 + wave * 16;
    const int colOrig = blockIdx.y * COLS_PER_SPLIT;
    const int rowBaseS = __builtin_amdgcn_readfirstlane(rowBase);  // SGPR

    // ---- A fragment: 16 rows x 128 K, loaded once (ISA 16-bit A layout) ----
    v16h a[4];
    {
        const _Float16* arow = zn + (size_t)(rowBase + lm) * DDIM;
#pragma unroll
        for (int kc = 0; kc < 4; ++kc) {
            const int kb = kc * 32;
            v8h lo = *(const v8h*)(arow + kb + halfsel * 8);
            v8h hi = *(const v8h*)(arow + kb + 16 + halfsel * 8);
            a[kc] = cat8(lo, hi);
        }
    }

    float rowsum[8];
    float pnum[8];
#pragma unroll
    for (int v = 0; v < 8; ++v) { rowsum[v] = 0.0f; pnum[v] = 0.0f; }

    const int nPanels = COLS_PER_SPLIT / PANEL;   // 8
    for (int pidx = 0; pidx < nPanels; ++pidx) {
        const int colPanel = colOrig + pidx * PANEL;

        // Cooperative panel load: 128 rows x 128 halves, contiguous in zn.
        {
            const v8h* gsrc = (const v8h*)(zn + (size_t)colPanel * DDIM);
            v8h*       ldst = (v8h*)Bs;
            const int  nvec = (PANEL * DDIM) / 8;          // 2048 v8h
#pragma unroll
            for (int i = 0; i < nvec / 256; ++i)           // 8 per thread
                ldst[threadIdx.x + i * 256] = gsrc[threadIdx.x + i * 256];
        }
        // Prefetch next panel into cache while we compute this one.
        if (pidx + 1 < nPanels) {
            const char* np = (const char*)(zn + (size_t)(colPanel + PANEL) * DDIM);
            __builtin_prefetch(np + threadIdx.x * 128, 0, 1);
        }
        __syncthreads();

#pragma unroll 1
        for (int ct = 0; ct < PANEL / 16; ++ct) {          // 8 sub-tiles
            const int colBase = colPanel + ct * 16;

            // Preload ALL four B fragments first so the 8 ds_load_b128 can be
            // in flight together and overlap the WMMA chain.
            const _Float16* brow = Bs + (size_t)(ct * 16 + lm) * DDIM;
            v16h b[4];
#pragma unroll
            for (int kc = 0; kc < 4; ++kc) {
                const int kb = kc * 32;
                v8h lo = *(const v8h*)(brow + kb + halfsel * 8);
                v8h hi = *(const v8h*)(brow + kb + 16 + halfsel * 8);
                b[kc] = cat8(lo, hi);
            }

            v8f c = {};
#pragma unroll
            for (int kc = 0; kc < 4; ++kc)
                c = __builtin_amdgcn_wmma_f32_16x16x32_f16(
                        false, a[kc], false, b[kc], (short)0, c, false, false);

            // Post-process: row = rowBase + 8*halfsel + v, col = colBase + lm.
            // Scalar (SGPR) compare -> s_cmp/s_cbranch, no exec-mask dance.
            if (__builtin_expect(colBase != rowBaseS, 1)) {
                // Hot path: no diagonal, no partner in this tile.
#pragma unroll
                for (int v = 0; v < 8; ++v)
                    rowsum[v] += __expf(2.0f * c[v]);
            } else {
                // Exactly one tile per row sweep: handle diag + partner.
                const int gcol = colBase + lm;
#pragma unroll
                for (int v = 0; v < 8; ++v) {
                    const int grow = rowBase + halfsel * 8 + v;
                    const float s  = c[v];
                    const float e  = __expf(2.0f * s);      // exp(s / 0.5)
                    rowsum[v] += (gcol == grow) ? 0.0f : e; // drop diagonal
                    pnum[v]   += (gcol == (grow ^ 1)) ? s : 0.0f;
                }
            }
        }
        __syncthreads();
    }

    // Reduce over the 16 lanes that share each row (xor masks < 16 stay in half)
#pragma unroll
    for (int v = 0; v < 8; ++v) {
#pragma unroll
        for (int m = 8; m >= 1; m >>= 1) {
            rowsum[v] += __shfl_xor(rowsum[v], m, 32);
            pnum[v]   += __shfl_xor(pnum[v],   m, 32);
        }
    }
    if (lm == 0) {
#pragma unroll
        for (int v = 0; v < 8; ++v) {
            const int grow = rowBase + halfsel * 8 + v;
            atomicAdd(&den[grow], rowsum[v]);
            atomicAdd(&num[grow], pnum[v]);
        }
    }
}

// ---------------------------------------------------------------------------
// Kernel 3: loss_r = log(den_r) - 2*num_r; mean over 8192 rows into out[0].
// ---------------------------------------------------------------------------
__launch_bounds__(256)
__global__ void finalize(const float* __restrict__ den,
                         const float* __restrict__ num,
                         float* __restrict__ out) {
    const int r = blockIdx.x * 256 + threadIdx.x;
    float term = logf(den[r]) - 2.0f * num[r];
#pragma unroll
    for (int m = 16; m >= 1; m >>= 1) term += __shfl_xor(term, m, 32);

    __shared__ float wsum[8];
    const int wave = threadIdx.x >> 5;
    const int lane = threadIdx.x & 31;
    if (lane == 0) wsum[wave] = term;
    __syncthreads();
    if (threadIdx.x == 0) {
        float t = 0.0f;
#pragma unroll
        for (int i = 0; i < 8; ++i) t += wsum[i];
        atomicAdd(out, t * (1.0f / (float)NROWS));
    }
}

// ---------------------------------------------------------------------------
extern "C" void kernel_launch(void* const* d_in, const int* in_sizes, int n_in,
                              void* d_out, int out_size, void* d_ws, size_t ws_size,
                              hipStream_t stream) {
    const float* zi = (const float*)d_in[0];
    const float* zj = (const float*)d_in[1];
    float* out = (float*)d_out;

    // Workspace layout: Zn f16 [8192*128] (2 MB) | den f32[8192] | num f32[8192]
    _Float16* zn = (_Float16*)d_ws;
    char* base   = (char*)d_ws;
    float* den   = (float*)(base + (size_t)NROWS * DDIM * sizeof(_Float16));
    float* num   = den + NROWS;

    hipMemsetAsync(den, 0, 2 * NROWS * sizeof(float), stream);
    hipMemsetAsync(out, 0, sizeof(float), stream);

    normalize_rows<<<NROWS / 8, 256, 0, stream>>>(zi, zj, zn);

    dim3 grid(NROWS / 128, NSPLIT);
    gram_loss<<<grid, 256, 0, stream>>>(zn, den, num);

    finalize<<<NROWS / 256, 256, 0, stream>>>(den, num, out);
}
